// MDCT_19464791785521
// MI455X (gfx1250) — matrix operand order repaired and checked by reading, not compiled, not run on hardware
//
#include <hip/hip_runtime.h>

// ---------------------------------------------------------------------------
// MDCT as batched GEMM on CDNA5 WMMA (bf16x3 split for ~fp32 accuracy).
//   out[b,k,f] = sum_t F[k,t] * x[b, (f-1)*512 + t],  zero padded.
// D(512 x 2049) = F(512 x 1024) @ Xframes(1024 x 2049), for b = 0..15.
// ---------------------------------------------------------------------------

typedef __attribute__((ext_vector_type(16))) __bf16 v16bf;
typedef __attribute__((ext_vector_type(8)))  __bf16 bf16x8;
typedef __attribute__((ext_vector_type(4)))  __bf16 bf16x4;
typedef __attribute__((ext_vector_type(8)))  float  v8f;
typedef __attribute__((ext_vector_type(4)))  float  f32x4;

#define MDCT_N    512
#define MDCT_T    2048
#define MDCT_NF   2049                 // T + 1 output frames
#define MDCT_NT   (MDCT_N * MDCT_T)    // signal length per batch
#define MDCT_B    16
#define KDIM      1024                 // 2N (filter taps)

#define BM 128                         // filter-row tile
#define BN 64                          // frame tile
#define BK 64                          // K slice staged per iteration
#define LDS_PITCH 136                  // 64 hi + 64 lo + 8 pad (bf16), 272B rows

// split f32 -> (hi, lo) bf16 pair:  v ~= hi + lo
__device__ __forceinline__ void split4(const f32x4 v, bf16x4& h, bf16x4& l) {
#pragma unroll
  for (int e = 0; e < 4; ++e) {
    __bf16 hh = (__bf16)v[e];
    h[e] = hh;
    l[e] = (__bf16)(v[e] - (float)hh);
  }
}

__device__ __forceinline__ v16bf cat8(bf16x8 a, bf16x8 b) {
  return __builtin_shufflevector(a, b, 0, 1, 2, 3, 4, 5, 6, 7,
                                        8, 9, 10, 11, 12, 13, 14, 15);
}

__global__ __launch_bounds__(256)
void mdct_wmma_bf16x3_kernel(const float* __restrict__ x,
                             const float* __restrict__ flt,
                             float* __restrict__ out) {
  __shared__ __align__(16) __bf16 sA[BM * LDS_PITCH];  // filter tile, hi|lo
  __shared__ __align__(16) __bf16 sB[BN * LDS_PITCH];  // frame  tile, hi|lo

  const int tid  = threadIdx.x;
  const int lane = tid & 31;
  const int wid  = tid >> 5;     // 0..7
  const int wm   = wid & 3;      // wave M coord (4)
  const int wn   = wid >> 2;     // wave N coord (2)
  const int lh   = lane >> 4;    // half-wave select
  const int l15  = lane & 15;

  const int n0 = blockIdx.x * BN;   // frame-tile origin
  const int m0 = blockIdx.y * BM;   // filter-row tile origin
  const int b  = blockIdx.z;        // batch

  const float* xb = x + (size_t)b * MDCT_NT;

  v8f acc[2][2];
#pragma unroll
  for (int i = 0; i < 2; ++i)
#pragma unroll
    for (int j = 0; j < 2; ++j) acc[i][j] = {};

  for (int kt = 0; kt < KDIM; kt += BK) {
    // ---- stage A: filter[m0..m0+127][kt..kt+63], split hi/lo ------------
#pragma unroll
    for (int i = 0; i < 8; ++i) {
      const int idx = i * 256 + tid;        // 0..2047
      const int r   = idx >> 4;             // 0..127
      const int c4  = idx & 15;             // float4 column
      const float* gp = flt + (size_t)(m0 + r) * KDIM + kt + c4 * 4;
      f32x4 v = *(const f32x4*)gp;
      if (kt + BK < KDIM) __builtin_prefetch(gp + BK, 0, 0);
      bf16x4 h, l;
      split4(v, h, l);
      *(bf16x4*)(&sA[r * LDS_PITCH + c4 * 4])      = h;
      *(bf16x4*)(&sA[r * LDS_PITCH + 64 + c4 * 4]) = l;
    }
    // ---- stage B: frames n0..n0+63, taps kt..kt+63, with zero padding ---
#pragma unroll
    for (int i = 0; i < 4; ++i) {
      const int idx = i * 256 + tid;        // 0..1023
      const int fr  = idx >> 4;             // frame within tile
      const int c4  = idx & 15;
      const int f   = n0 + fr;
      const int k   = kt + c4 * 4;
      const int g   = (f - 1) * MDCT_N + k; // sample index (mult of 4)
      f32x4 v = {};
      if (f < MDCT_NF && g >= 0 && g < MDCT_NT) {
        v = *(const f32x4*)(xb + g);
        if (kt + BK < KDIM && g + BK < MDCT_NT)
          __builtin_prefetch(xb + g + BK, 0, 0);
      }
      bf16x4 h, l;
      split4(v, h, l);
      *(bf16x4*)(&sB[fr * LDS_PITCH + c4 * 4])      = h;
      *(bf16x4*)(&sB[fr * LDS_PITCH + 64 + c4 * 4]) = l;
    }
    __syncthreads();

    // ---- compute: two K=32 steps over the staged slice ------------------
#pragma unroll
    for (int ks = 0; ks < BK; ks += 32) {
      v16bf ah[2], al[2], bh[2], bl[2];
#pragma unroll
      for (int ti = 0; ti < 2; ++ti) {
        // A 16x32 layout: lanes 0-15 row=l15 K={0..7,16..23}; lanes 16-31 K={8..15,24..31}
        const __bf16* base = &sA[(wm * 32 + ti * 16 + l15) * LDS_PITCH];
        const int ka = ks + lh * 8;
        ah[ti] = cat8(*(const bf16x8*)(base + ka),
                      *(const bf16x8*)(base + ka + 16));
        al[ti] = cat8(*(const bf16x8*)(base + 64 + ka),
                      *(const bf16x8*)(base + 64 + ka + 16));
      }
#pragma unroll
      for (int tj = 0; tj < 2; ++tj) {
        // B 32x16 layout: lane holds one column; lanes 0-15 K={0..7,16..23}, 16-31 K={8..15,24..31}
        const __bf16* base = &sB[(wn * 32 + tj * 16 + l15) * LDS_PITCH];
        const int kb = ks + lh * 8;
        bh[tj] = cat8(*(const bf16x8*)(base + kb),
                      *(const bf16x8*)(base + kb + 16));
        bl[tj] = cat8(*(const bf16x8*)(base + 64 + kb),
                      *(const bf16x8*)(base + 64 + kb + 16));
      }
      // three passes of 4 independent WMMAs each (covers RAW hazard window)
#pragma unroll
      for (int ti = 0; ti < 2; ++ti)
#pragma unroll
        for (int tj = 0; tj < 2; ++tj)
          acc[ti][tj] = __builtin_amdgcn_wmma_f32_16x16x32_bf16(
              false, ah[ti], false, bh[tj], (short)0, acc[ti][tj], false, false);
#pragma unroll
      for (int ti = 0; ti < 2; ++ti)
#pragma unroll
        for (int tj = 0; tj < 2; ++tj)
          acc[ti][tj] = __builtin_amdgcn_wmma_f32_16x16x32_bf16(
              false, ah[ti], false, bl[tj], (short)0, acc[ti][tj], false, false);
#pragma unroll
      for (int ti = 0; ti < 2; ++ti)
#pragma unroll
        for (int tj = 0; tj < 2; ++tj)
          acc[ti][tj] = __builtin_amdgcn_wmma_f32_16x16x32_bf16(
              false, al[ti], false, bh[tj], (short)0, acc[ti][tj], false, false);
    }
    __syncthreads();
  }

  // ---- store: C/D 16x16 f32 layout: VGPR r -> M = r + 8*lh, N = l15 -----
#pragma unroll
  for (int ti = 0; ti < 2; ++ti)
#pragma unroll
    for (int tj = 0; tj < 2; ++tj) {
      const int col = n0 + wn * 32 + tj * 16 + l15;
      if (col < MDCT_NF) {
        const int rbase = m0 + wm * 32 + ti * 16 + 8 * lh;
        float* o = out + ((size_t)b * MDCT_N + rbase) * MDCT_NF + col;
#pragma unroll
        for (int r = 0; r < 8; ++r)
          o[(size_t)r * MDCT_NF] = acc[ti][tj][r];
      }
    }
}

extern "C" void kernel_launch(void* const* d_in, const int* in_sizes, int n_in,
                              void* d_out, int out_size, void* d_ws, size_t ws_size,
                              hipStream_t stream) {
  (void)in_sizes; (void)n_in; (void)out_size; (void)d_ws; (void)ws_size;
  const float* x   = (const float*)d_in[0];   // [16, 1, 512*2048] f32
  const float* flt = (const float*)d_in[1];   // [512, 1, 1024]    f32
  float* out = (float*)d_out;                 // [16, 512, 2049]   f32

  dim3 grid((MDCT_NF + BN - 1) / BN,          // 33 frame tiles
            MDCT_N / BM,                      // 4 row tiles
            MDCT_B);                          // 16 batches
  dim3 block(256);
  mdct_wmma_bf16x3_kernel<<<grid, block, 0, stream>>>(x, flt, out);
}
